// XLRelMultiHeadedSelfAttention_26079041421960
// MI455X (gfx1250) — matrix operand order repaired and checked by reading, not compiled, not run on hardware
//
#include <hip/hip_runtime.h>
#include <hip/hip_bf16.h>
#include <stdint.h>

#define BB 4
#define TT 1024
#define DD 512
#define NH 8
#define DKK 64
#define SSS 2047
#define BT (BB*TT)
#define BTD (BB*TT*DD)

typedef __attribute__((ext_vector_type(16))) __bf16 v16bf;
typedef __attribute__((ext_vector_type(8)))  float  v8f;

struct alignas(16) U4 { unsigned int a, b, c, d; };
struct alignas(16) F4 { float v[4]; };

__device__ __forceinline__ v8f wmma_bf16(v16bf av, v16bf bv, v8f cv) {
  return __builtin_amdgcn_wmma_f32_16x16x32_bf16(false, av, false, bv, (short)0, cv, false, false);
}

__device__ __forceinline__ v8f v8f_zero() {
  v8f z;
#pragma unroll
  for (int i = 0; i < 8; ++i) z[i] = 0.0f;
  return z;
}

// A-matrix fragment (16x32 bf16), lane = M (mod 16).
// half==0: K = ck+{0..15};  half==1: K = ck+{16..31}
__device__ __forceinline__ v16bf load_afrag(const __bf16* row, int ck, int half) {
  union { v16bf v; U4 q[2]; } f;
  f.q[0] = *reinterpret_cast<const U4*>(row + ck + half * 16);
  f.q[1] = *reinterpret_cast<const U4*>(row + ck + half * 16 + 8);
  return f.v;
}

// B-matrix fragment (32x16 bf16), lane = N (mod 16); 16 contiguous K values.
__device__ __forceinline__ v16bf load_bfrag(const __bf16* p) {
  union { v16bf v; U4 q[2]; } f;
  f.q[0] = *reinterpret_cast<const U4*>(p);
  f.q[1] = *reinterpret_cast<const U4*>(p + 8);
  return f.v;
}

// ---------------- elementwise kernels ----------------

__global__ void cvt_bf16_kernel(const float* __restrict__ in, __bf16* __restrict__ out, int n) {
  int i = blockIdx.x * 256 + threadIdx.x;
  if (i < n) out[i] = (__bf16)in[i];
}

// W [K, Nc] f32 row-major -> WT [Nc, K] bf16 row-major
__global__ void trans_bf16_kernel(const float* __restrict__ W, __bf16* __restrict__ WT, int K, int Nc) {
  int i = blockIdx.x * 256 + threadIdx.x;
  if (i < K * Nc) {
    int k = i / Nc;
    int c = i % Nc;
    WT[(size_t)c * K + k] = (__bf16)W[i];
  }
}

// q f32 [B,T,D] -> qu/qv bf16 with posu/posv [D] broadcast over (b,t)
__global__ void qpost_kernel(const float* __restrict__ q, const float* __restrict__ posu,
                             const float* __restrict__ posv, __bf16* __restrict__ qu,
                             __bf16* __restrict__ qv, int total) {
  int i = blockIdx.x * 256 + threadIdx.x;
  if (i < total) {
    int d = i % DD;
    float qq = q[i];
    qu[i] = (__bf16)(qq + posu[d]);
    qv[i] = (__bf16)(qq + posv[d]);
  }
}

// kv f32 [B,T,2D] -> kb bf16 [B,T,D] and vT bf16 [B,N,DK,T]
__global__ void kvpost_kernel(const float* __restrict__ kv, __bf16* __restrict__ kb,
                              __bf16* __restrict__ vT, int total) {
  int i = blockIdx.x * 256 + threadIdx.x;
  if (i < total) {
    int b = i / (TT * DD);
    int rem = i % (TT * DD);
    int t = rem / DD;
    int d = rem % DD;
    size_t kvrow = ((size_t)b * TT + t) * (2 * DD);
    kb[i] = (__bf16)kv[kvrow + d];
    float vv = kv[kvrow + DD + d];
    int n = d / DKK, dk = d % DKK;
    vT[(((size_t)(b * NH + n)) * DKK + dk) * TT + t] = (__bf16)vv;
  }
}

// ---------------- generic WMMA GEMM: C[M,Nc] = A[M,512](bf16) @ WT^T + bias ----------------
// WT stored [Nc, 512]. One wave computes a 32x64 tile, ping-pong software pipeline over K.
__global__ __launch_bounds__(32, 1)
void gemm_bias_kernel(const __bf16* __restrict__ A, const __bf16* __restrict__ WT,
                      const float* __restrict__ bias, float* __restrict__ C,
                      int M, int Nc) {
  int ntiles_n = Nc / 64;
  int tile = blockIdx.x;
  int i0 = (tile / ntiles_n) * 32;
  int nc0 = (tile % ntiles_n) * 64;
  int lane = threadIdx.x;
  int half = lane >> 4;
  int lm = lane & 15;

  int ar0 = i0 + lm;      if (ar0 >= M) ar0 = M - 1;
  int ar1 = i0 + 16 + lm; if (ar1 >= M) ar1 = M - 1;
  const __bf16* A0 = A + (size_t)ar0 * DD;
  const __bf16* A1 = A + (size_t)ar1 * DD;
  const __bf16* Wb = WT + (size_t)(nc0 + lm) * DD + half * 16;

  v8f acc[8];
#pragma unroll
  for (int t = 0; t < 8; ++t) acc[t] = v8f_zero();

  auto ld = [&](int ck, v16bf& a0, v16bf& a1, v16bf& b0, v16bf& b1, v16bf& b2, v16bf& b3) {
    a0 = load_afrag(A0, ck, half);
    a1 = load_afrag(A1, ck, half);
    b0 = load_bfrag(Wb + 0 * (16 * DD) + ck);
    b1 = load_bfrag(Wb + 1 * (16 * DD) + ck);
    b2 = load_bfrag(Wb + 2 * (16 * DD) + ck);
    b3 = load_bfrag(Wb + 3 * (16 * DD) + ck);
  };
  auto mm = [&](v16bf a0, v16bf a1, v16bf b0, v16bf b1, v16bf b2, v16bf b3) {
    acc[0] = wmma_bf16(a0, b0, acc[0]);  acc[4] = wmma_bf16(a1, b0, acc[4]);
    acc[1] = wmma_bf16(a0, b1, acc[1]);  acc[5] = wmma_bf16(a1, b1, acc[5]);
    acc[2] = wmma_bf16(a0, b2, acc[2]);  acc[6] = wmma_bf16(a1, b2, acc[6]);
    acc[3] = wmma_bf16(a0, b3, acc[3]);  acc[7] = wmma_bf16(a1, b3, acc[7]);
  };

  // two named buffer sets (structural ping-pong, no register copies)
  v16bf xa0, xa1, xb0, xb1, xb2, xb3;
  v16bf ya0, ya1, yb0, yb1, yb2, yb3;

  ld(0, xa0, xa1, xb0, xb1, xb2, xb3);
  for (int c = 0; c < 7; ++c) {          // chunk pairs (64 K each); chunks 0..448
    int ck = c * 64;
    ld(ck + 32, ya0, ya1, yb0, yb1, yb2, yb3);
    mm(xa0, xa1, xb0, xb1, xb2, xb3);    // chunk ck
    ld(ck + 64, xa0, xa1, xb0, xb1, xb2, xb3);
    mm(ya0, ya1, yb0, yb1, yb2, yb3);    // chunk ck+32
  }
  ld(480, ya0, ya1, yb0, yb1, yb2, yb3);
  mm(xa0, xa1, xb0, xb1, xb2, xb3);      // chunk 448
  mm(ya0, ya1, yb0, yb1, yb2, yb3);      // chunk 480

#pragma unroll
  for (int t = 0; t < 4; ++t) {
    int col = nc0 + t * 16 + lm;
    float bv = bias ? bias[col] : 0.0f;
#pragma unroll
    for (int rt = 0; rt < 2; ++rt) {
#pragma unroll
      for (int r = 0; r < 8; ++r) {
        int row = i0 + rt * 16 + r + half * 8;
        if (row < M) C[(size_t)row * Nc + col] = acc[4 * rt + t][r] + bv;
      }
    }
  }
}

// ---------------- scores: AC + shifted BD, masked & scaled -> weights ----------------
__global__ __launch_bounds__(32, 1)
void scores_kernel(const __bf16* __restrict__ qu, const __bf16* __restrict__ qv,
                   const __bf16* __restrict__ kmat, const __bf16* __restrict__ pb,
                   const unsigned char* __restrict__ mask, float* __restrict__ weights) {
  __shared__ float mbd[16 * 32];
  int blk = blockIdx.x;                 // b*N*(T/16) blocks
  int i0 = (blk % (TT / 16)) * 16;
  int bn = blk / (TT / 16);
  int n = bn % NH;
  int b = bn / NH;
  int lane = threadIdx.x;
  int half = lane >> 4;
  int lm = lane & 15;

  const __bf16* qurow = qu + ((size_t)(b * TT + i0 + lm) * DD + n * DKK);
  const __bf16* qvrow = qv + ((size_t)(b * TT + i0 + lm) * DD + n * DKK);
  v16bf au0 = load_afrag(qurow, 0, half);
  v16bf au1 = load_afrag(qurow, 32, half);
  v16bf av0 = load_afrag(qvrow, 0, half);
  v16bf av1 = load_afrag(qvrow, 32, half);

  const __bf16* kbase = kmat + ((size_t)(b * TT + lm) * DD + n * DKK) + half * 16;

  auto load_tile = [&](int j0, v16bf& k0, v16bf& k1,
                       v16bf& p00, v16bf& p01, v16bf& p10, v16bf& p11) {
    const __bf16* krow = kbase + (size_t)j0 * DD;
    k0 = load_bfrag(krow);
    k1 = load_bfrag(krow + 32);
    int baser = (TT - 1) + j0 - i0 - 15;         // always >= 0
    int pr0 = baser + lm;      if (pr0 > SSS - 1) pr0 = SSS - 1;
    int pr1 = baser + 16 + lm; if (pr1 > SSS - 1) pr1 = SSS - 1;
    const __bf16* prow0 = pb + ((size_t)pr0 * DD + n * DKK) + half * 16;
    const __bf16* prow1 = pb + ((size_t)pr1 * DD + n * DKK) + half * 16;
    p00 = load_bfrag(prow0);
    p01 = load_bfrag(prow0 + 32);
    p10 = load_bfrag(prow1);
    p11 = load_bfrag(prow1 + 32);
  };

  auto compute_tile = [&](int j0, v16bf k0, v16bf k1,
                          v16bf p00, v16bf p01, v16bf p10, v16bf p11) {
    // AC tile: q_u (16x64) x k^T (64x16)
    v8f acc = v8f_zero();
    acc = wmma_bf16(au0, k0, acc);
    acc = wmma_bf16(au1, k1, acc);
    // matrix_bd sub-tile: 16 rows x 32 p-rows
    v8f bd0 = v8f_zero();
    v8f bd1 = v8f_zero();
    bd0 = wmma_bf16(av0, p00, bd0);
    bd0 = wmma_bf16(av1, p01, bd0);
    bd1 = wmma_bf16(av0, p10, bd1);
    bd1 = wmma_bf16(av1, p11, bd1);

    __syncthreads();   // protect prior tile's mbd reads
#pragma unroll
    for (int r = 0; r < 8; ++r) {
      int ii = r + half * 8;
      mbd[ii * 32 + lm] = bd0[r];
      mbd[ii * 32 + 16 + lm] = bd1[r];
    }
    __syncthreads();

#pragma unroll
    for (int r = 0; r < 8; ++r) {
      int ii = r + half * 8;
      int jj = lm;
      float s = (acc[r] + mbd[ii * 32 + (jj - ii + 15)]) * 0.125f;  // /sqrt(64)
      int gi = i0 + ii, gj = j0 + jj;
      unsigned char m = mask[((size_t)b * TT + gi) * TT + gj];
      if (!m) s = -100000.0f;
      weights[((size_t)(b * NH + n) * TT + gi) * TT + gj] = s;
    }
  };

  // structural ping-pong: buffers X and Y
  v16bf xk0, xk1, xp0, xp1, xp2, xp3;
  v16bf yk0, yk1, yp0, yp1, yp2, yp3;

  load_tile(0, xk0, xk1, xp0, xp1, xp2, xp3);
  for (int j0 = 0; j0 < TT; j0 += 32) {
    load_tile((j0 + 16) & (TT - 1), yk0, yk1, yp0, yp1, yp2, yp3);
    compute_tile(j0, xk0, xk1, xp0, xp1, xp2, xp3);
    load_tile((j0 + 32) & (TT - 1), xk0, xk1, xp0, xp1, xp2, xp3);
    compute_tile(j0 + 16, yk0, yk1, yp0, yp1, yp2, yp3);
  }
}

// ---------------- row softmax, in place ----------------
__global__ void softmax_kernel(float* __restrict__ weights) {
  __shared__ float red[256];
  int row = blockIdx.x;                 // B*N*T rows of length T
  float* w = weights + (size_t)row * TT;
  int tid = threadIdx.x;
  float x[4];
  float m = -1e30f;
#pragma unroll
  for (int i = 0; i < 4; ++i) { x[i] = w[tid + i * 256]; m = fmaxf(m, x[i]); }
  red[tid] = m; __syncthreads();
  for (int s = 128; s > 0; s >>= 1) {
    if (tid < s) red[tid] = fmaxf(red[tid], red[tid + s]);
    __syncthreads();
  }
  m = red[0]; __syncthreads();
  float sum = 0.0f;
#pragma unroll
  for (int i = 0; i < 4; ++i) { x[i] = __expf(x[i] - m); sum += x[i]; }
  red[tid] = sum; __syncthreads();
  for (int s = 128; s > 0; s >>= 1) {
    if (tid < s) red[tid] += red[tid + s];
    __syncthreads();
  }
  float inv = 1.0f / red[0];
#pragma unroll
  for (int i = 0; i < 4; ++i) w[tid + i * 256] = x[i] * inv;
}

// ---------------- context = weights @ v  -> ctx bf16 [B,T,D] ----------------
__global__ __launch_bounds__(32, 1)
void context_kernel(const float* __restrict__ weights, const __bf16* __restrict__ vT,
                    __bf16* __restrict__ ctx) {
  int blk = blockIdx.x;
  int i0 = (blk % (TT / 16)) * 16;
  int bn = blk / (TT / 16);
  int n = bn % NH;
  int b = bn / NH;
  int lane = threadIdx.x;
  int half = lane >> 4;
  int lm = lane & 15;

  const float* wrow = weights + ((size_t)(b * NH + n) * TT + i0 + lm) * TT;
  const __bf16* vbase = vT + (size_t)(b * NH + n) * DKK * TT + (size_t)lm * TT + half * 16;

  v8f acc[4];
#pragma unroll
  for (int t = 0; t < 4; ++t) acc[t] = v8f_zero();

  auto load_chunk = [&](int ck, F4& w0, F4& w1, F4& w2, F4& w3,
                        v16bf& b0, v16bf& b1, v16bf& b2, v16bf& b3) {
    const float* p0 = wrow + ck + half * 16;
    const float* p1 = wrow + ck + half * 16 + 8;
    w0 = *reinterpret_cast<const F4*>(p0);
    w1 = *reinterpret_cast<const F4*>(p0 + 4);
    w2 = *reinterpret_cast<const F4*>(p1);
    w3 = *reinterpret_cast<const F4*>(p1 + 4);
    b0 = load_bfrag(vbase + 0 * (16 * TT) + ck);
    b1 = load_bfrag(vbase + 1 * (16 * TT) + ck);
    b2 = load_bfrag(vbase + 2 * (16 * TT) + ck);
    b3 = load_bfrag(vbase + 3 * (16 * TT) + ck);
  };
  auto compute_chunk = [&](F4 w0, F4 w1, F4 w2, F4 w3,
                           v16bf b0, v16bf b1, v16bf b2, v16bf b3) {
    union { v16bf v; __bf16 h[16]; } af;
#pragma unroll
    for (int e = 0; e < 4; ++e) {
      af.h[e]      = (__bf16)w0.v[e];
      af.h[4 + e]  = (__bf16)w1.v[e];
      af.h[8 + e]  = (__bf16)w2.v[e];
      af.h[12 + e] = (__bf16)w3.v[e];
    }
    acc[0] = wmma_bf16(af.v, b0, acc[0]);
    acc[1] = wmma_bf16(af.v, b1, acc[1]);
    acc[2] = wmma_bf16(af.v, b2, acc[2]);
    acc[3] = wmma_bf16(af.v, b3, acc[3]);
  };

  F4 xw0, xw1, xw2, xw3;  v16bf xb0, xb1, xb2, xb3;
  F4 yw0, yw1, yw2, yw3;  v16bf yb0, yb1, yb2, yb3;

  load_chunk(0, xw0, xw1, xw2, xw3, xb0, xb1, xb2, xb3);
  for (int ck = 0; ck < TT; ck += 64) {
    load_chunk((ck + 32) & (TT - 1), yw0, yw1, yw2, yw3, yb0, yb1, yb2, yb3);
    compute_chunk(xw0, xw1, xw2, xw3, xb0, xb1, xb2, xb3);
    load_chunk((ck + 64) & (TT - 1), xw0, xw1, xw2, xw3, xb0, xb1, xb2, xb3);
    compute_chunk(yw0, yw1, yw2, yw3, yb0, yb1, yb2, yb3);
  }

#pragma unroll
  for (int t = 0; t < 4; ++t) {
#pragma unroll
    for (int r = 0; r < 8; ++r) {
      int row = i0 + r + half * 8;
      int col = n * DKK + t * 16 + lm;
      ctx[(size_t)(b * TT + row) * DD + col] = (__bf16)acc[t][r];
    }
  }
}

// ---------------- host launcher ----------------
extern "C" void kernel_launch(void* const* d_in, const int* in_sizes, int n_in,
                              void* d_out, int out_size, void* d_ws, size_t ws_size,
                              hipStream_t stream) {
  (void)in_sizes; (void)n_in; (void)out_size; (void)ws_size;

  const float* x    = (const float*)d_in[0];
  const float* x1   = (const float*)d_in[1];
  const unsigned char* mask = (const unsigned char*)d_in[2];
  const float* pos  = (const float*)d_in[3];
  const float* Wq   = (const float*)d_in[4];
  const float* bq   = (const float*)d_in[5];
  const float* Wvk  = (const float*)d_in[6];
  const float* bvk  = (const float*)d_in[7];
  const float* Wpos = (const float*)d_in[8];
  const float* posu = (const float*)d_in[9];
  const float* posv = (const float*)d_in[10];
  const float* Wo   = (const float*)d_in[11];
  const float* bo   = (const float*)d_in[12];

  float* out_ptr = (float*)d_out;                 // [B,T,D]
  float* weights = out_ptr + (size_t)BTD;         // [B,N,T,T]

  char* base = (char*)d_ws;
  size_t off = 0;
  auto alloc = [&](size_t bytes) -> char* {
    char* p = base + off;
    off += (bytes + 255) & ~(size_t)255;
    return p;
  };

  __bf16* xb    = (__bf16*)alloc((size_t)BTD * 2);
  __bf16* x1b   = (__bf16*)alloc((size_t)BTD * 2);
  __bf16* posb  = (__bf16*)alloc((size_t)SSS * DD * 2);
  __bf16* WqT   = (__bf16*)alloc((size_t)DD * DD * 2);
  __bf16* WvkT  = (__bf16*)alloc((size_t)DD * 2 * DD * 2);
  __bf16* WposT = (__bf16*)alloc((size_t)DD * DD * 2);
  __bf16* WoT   = (__bf16*)alloc((size_t)DD * DD * 2);
  float*  qf    = (float*)alloc((size_t)BTD * 4);
  float*  kvf   = (float*)alloc((size_t)BT * 2 * DD * 4);
  float*  pf    = (float*)alloc((size_t)SSS * DD * 4);
  __bf16* qub   = (__bf16*)alloc((size_t)BTD * 2);
  __bf16* qvb   = (__bf16*)alloc((size_t)BTD * 2);
  __bf16* kbb   = (__bf16*)alloc((size_t)BTD * 2);
  __bf16* vTb   = (__bf16*)alloc((size_t)BTD * 2);
  __bf16* pbb   = (__bf16*)alloc((size_t)SSS * DD * 2);
  __bf16* ctxb  = (__bf16*)alloc((size_t)BTD * 2);

  // 1) convert activations / weights to bf16 (weights transposed to [Nc,K])
  cvt_bf16_kernel<<<(BTD + 255) / 256, 256, 0, stream>>>(x, xb, BTD);
  cvt_bf16_kernel<<<(BTD + 255) / 256, 256, 0, stream>>>(x1, x1b, BTD);
  cvt_bf16_kernel<<<(SSS * DD + 255) / 256, 256, 0, stream>>>(pos, posb, SSS * DD);
  trans_bf16_kernel<<<(DD * DD + 255) / 256, 256, 0, stream>>>(Wq, WqT, DD, DD);
  trans_bf16_kernel<<<(DD * 2 * DD + 255) / 256, 256, 0, stream>>>(Wvk, WvkT, DD, 2 * DD);
  trans_bf16_kernel<<<(DD * DD + 255) / 256, 256, 0, stream>>>(Wpos, WposT, DD, DD);
  trans_bf16_kernel<<<(DD * DD + 255) / 256, 256, 0, stream>>>(Wo, WoT, DD, DD);

  // 2) projection GEMMs (WMMA bf16, f32 accumulate), 32x64 tiles
  gemm_bias_kernel<<<(BT / 32) * (DD / 64), 32, 0, stream>>>(xb, WqT, bq, qf, BT, DD);
  gemm_bias_kernel<<<(BT / 32) * (2 * DD / 64), 32, 0, stream>>>(x1b, WvkT, bvk, kvf, BT, 2 * DD);
  gemm_bias_kernel<<<((SSS + 31) / 32) * (DD / 64), 32, 0, stream>>>(posb, WposT, nullptr, pf, SSS, DD);

  // 3) post-processing into WMMA-friendly bf16 layouts
  qpost_kernel<<<(BTD + 255) / 256, 256, 0, stream>>>(qf, posu, posv, qub, qvb, BTD);
  kvpost_kernel<<<(BTD + 255) / 256, 256, 0, stream>>>(kvf, kbb, vTb, BTD);
  cvt_bf16_kernel<<<(SSS * DD + 255) / 256, 256, 0, stream>>>(pf, pbb, SSS * DD);

  // 4) scores (AC + shifted BD) -> weights, softmax in place
  scores_kernel<<<BB * NH * (TT / 16), 32, 0, stream>>>(qub, qvb, kbb, pbb, mask, weights);
  softmax_kernel<<<BB * NH * TT, 256, 0, stream>>>(weights);

  // 5) context = weights @ v, then output projection
  context_kernel<<<BB * NH * (TT / 16), 32, 0, stream>>>(weights, vTb, ctxb);
  gemm_bias_kernel<<<(BT / 32) * (DD / 64), 32, 0, stream>>>(ctxb, WoT, bo, out_ptr, BT, DD);
}
